// ExpertGroup_6219112645453
// MI455X (gfx1250) — compile-verified, matrix operand order checked by reading, not run on hardware
//
#include <hip/hip_runtime.h>
#include <hip/hip_bf16.h>

// Problem constants (match reference)
#define T_TOK 2048
#define NEXP  8
#define DDIM  1024
#define IDIM  2048
#define MAXMB 16   // 256-token M-blocks: T/256 + E = 16

typedef __attribute__((ext_vector_type(16))) _Float16 v16h;
typedef __attribute__((ext_vector_type(8)))  _Float16 v8h;
typedef __attribute__((ext_vector_type(8)))  float    v8f;
typedef __attribute__((ext_vector_type(4)))  unsigned int v4u;
typedef __attribute__((ext_vector_type(8)))  int      v8i;
typedef __attribute__((ext_vector_type(4)))  int      v4i;

// Workspace int layout (at base of d_ws):
//  [0]            nmb (number of active 256-token M-blocks)
//  [8  .. 23]     mb_expert
//  [24 .. 39]     mb_tok_start   (index into perm[])
//  [40 .. 55]     mb_tok_end     (expert's end index into perm[])
//  [64 .. 64+T)   perm (token ids grouped by expert)
//  byte 65536 ..  _Float16 hidden[T*I]  (8 MB)
#define WS_MB_E   8
#define WS_MB_S   24
#define WS_MB_END 40
#define WS_PERM   64
#define WS_HIDDEN_BYTE_OFF 65536

// ---------------------------------------------------------------------------
// TDM: DMA a 2D fp32 tile (tile_d0 contiguous elems x tile_d1 rows, row
// stride = stride0 elems) from global memory into LDS at lds_off.
// Descriptor layout per CDNA5 ISA 8.3-8.5 (D# group0 + group1, 2D tensor).
// 6-arg builtin: (uint32x4 g0, int32x8 g1, int32x4, int32x4, int32x8, i32 cpol)
// ---------------------------------------------------------------------------
__device__ __forceinline__ void tdm_load_2d(unsigned lds_off, const void* gptr,
                                            unsigned tensor_d0, unsigned tensor_d1,
                                            unsigned tile_d0, unsigned tile_d1,
                                            unsigned long long stride0_elems) {
  unsigned long long ga = (unsigned long long)(uintptr_t)gptr;
  v4u g0;
  g0[0] = 1u;                                             // count=1 (valid, user mode)
  g0[1] = lds_off;                                        // lds_addr (bytes)
  g0[2] = (unsigned)(ga & 0xFFFFFFFFu);                   // global_addr[31:0]
  g0[3] = (unsigned)((ga >> 32) & 0x1FFFFFFu) | (2u << 30); // addr[56:32] | type=2
  v8i g1;
  g1[0] = (int)(2u << 16);                                // data_size=4B; no mcast/pad
  g1[1] = (int)((tensor_d0 & 0xFFFFu) << 16);             // tensor_dim0[15:0]
  g1[2] = (int)(((tensor_d0 >> 16) & 0xFFFFu) |
                ((tensor_d1 & 0xFFFFu) << 16));           // dim0 hi | dim1 lo
  g1[3] = (int)(((tensor_d1 >> 16) & 0xFFFFu) |
                (tile_d0 << 16));                         // dim1 hi | tile_dim0
  g1[4] = (int)tile_d1;                                   // tile_dim1 (tile_dim2=0)
  g1[5] = (int)(unsigned)(stride0_elems & 0xFFFFFFFFu);   // tensor_dim0_stride lo
  g1[6] = (int)(unsigned)((stride0_elems >> 32) & 0xFFFFu);
  g1[7] = 0;
  v4i z4 = {0, 0, 0, 0};
  v8i z8 = {0, 0, 0, 0, 0, 0, 0, 0};
  __builtin_amdgcn_tensor_load_to_lds(g0, g1, z4, z4, z8, 0);
}

// ---------------------------------------------------------------------------
// Kernel 1: counting-sort tokens by expert + build 256-token M-block table.
// ---------------------------------------------------------------------------
__global__ __launch_bounds__(256) void moe_setup(const int* __restrict__ idx,
                                                 int* __restrict__ ws) {
  __shared__ int sc[NEXP];
  __shared__ int soff[NEXP + 1];
  __shared__ int scur[NEXP];
  const int tid = threadIdx.x;
  if (tid < NEXP) sc[tid] = 0;
  __syncthreads();
  for (int t = tid; t < T_TOK; t += 256) atomicAdd(&sc[idx[t]], 1);
  __syncthreads();
  if (tid == 0) {
    soff[0] = 0;
    for (int e = 0; e < NEXP; ++e) soff[e + 1] = soff[e] + sc[e];
    int nmb = 0;
    for (int e = 0; e < NEXP; ++e) {
      for (int b = 0; b < sc[e]; b += 256) {
        ws[WS_MB_E + nmb]   = e;
        ws[WS_MB_S + nmb]   = soff[e] + b;
        ws[WS_MB_END + nmb] = soff[e + 1];
        ++nmb;
      }
    }
    ws[0] = nmb;
    for (int e = 0; e < NEXP; ++e) scur[e] = soff[e];
  }
  __syncthreads();
  for (int t = tid; t < T_TOK; t += 256) {
    int e = idx[t];
    int p = atomicAdd(&scur[e], 1);
    ws[WS_PERM + p] = t;
  }
}

// Build one 16x32 A fragment (f16) from fp32 token rows (unconditional:
// padding slots alias a valid token; their results are never stored).
__device__ __forceinline__ v16h load_a_f32(const float* __restrict__ x,
                                           int tok, int kb, int ld) {
  v16h a;
  const float4* xp = (const float4*)(x + (size_t)tok * ld + kb);
  float4 f0 = xp[0], f1 = xp[1];   // K = kb .. kb+7
  float4 f2 = xp[4], f3 = xp[5];   // K = kb+16 .. kb+23
  a[0]  = (_Float16)f0.x; a[1]  = (_Float16)f0.y;
  a[2]  = (_Float16)f0.z; a[3]  = (_Float16)f0.w;
  a[4]  = (_Float16)f1.x; a[5]  = (_Float16)f1.y;
  a[6]  = (_Float16)f1.z; a[7]  = (_Float16)f1.w;
  a[8]  = (_Float16)f2.x; a[9]  = (_Float16)f2.y;
  a[10] = (_Float16)f2.z; a[11] = (_Float16)f2.w;
  a[12] = (_Float16)f3.x; a[13] = (_Float16)f3.y;
  a[14] = (_Float16)f3.z; a[15] = (_Float16)f3.w;
  return a;
}

// Build one 32x16 B fragment (f16) from an LDS-staged fp32 tile ([32 k][16 n]).
__device__ __forceinline__ v16h load_b_lds(const float* __restrict__ buf,
                                           int kadd, int rlo) {
  v16h b;
#pragma unroll
  for (int j = 0; j < 8; ++j) {
    b[j]     = (_Float16)buf[(kadd + j) * 16 + rlo];
    b[j + 8] = (_Float16)buf[(kadd + 16 + j) * 16 + rlo];
  }
  return b;
}

// ---------------------------------------------------------------------------
// Kernel 2: hidden = silu(x·Wg[e]) * (x·Wu[e]).
// Block = 4 waves x 64 tokens each (256 tokens) x 16 I-columns.
// Weight tiles staged to LDS by the Tensor Data Mover (wave 0 only),
// double-buffered and shared by all 4 waves.
// ---------------------------------------------------------------------------
__global__ __launch_bounds__(128) void moe_gemm1(const float* __restrict__ x,
                                                 const float* __restrict__ wg,
                                                 const float* __restrict__ wu,
                                                 const int* __restrict__ ws,
                                                 _Float16* __restrict__ hidden) {
  __shared__ float ldsg[2][32 * 16];
  __shared__ float ldsu[2][32 * 16];

  const int nmb = ws[0];
  if ((int)blockIdx.y >= nmb) return;
  const int e    = ws[WS_MB_E + blockIdx.y];
  const int ts   = ws[WS_MB_S + blockIdx.y];
  const int te   = ws[WS_MB_END + blockIdx.y];
  const int lane = threadIdx.x & 31;
  const int rlo  = lane & 15;
  const int kadd = (lane >> 4) << 3;          // 0 (lanes 0-15) / 8 (lanes 16-31)
  const int n0   = blockIdx.x * 16;
  const int n    = n0 + rlo;                  // I column
  const int wid  = __builtin_amdgcn_readfirstlane((int)(threadIdx.x >> 5));
  const int tsw  = ts + wid * 64;             // this wave's 64-token group

  const int tok0 = ws[WS_PERM + ts];          // fallback token for padding slots
  int tok[4];
#pragma unroll
  for (int mt = 0; mt < 4; ++mt) {
    int p = tsw + mt * 16 + rlo;
    tok[mt] = (p < te) ? ws[WS_PERM + p] : tok0;
  }

  v8f accg[4], accu[4];
#pragma unroll
  for (int mt = 0; mt < 4; ++mt)
#pragma unroll
    for (int j = 0; j < 8; ++j) { accg[mt][j] = 0.0f; accu[mt][j] = 0.0f; }

  const float* __restrict__ wgE = wg + (size_t)e * DDIM * IDIM + n0;
  const float* __restrict__ wuE = wu + (size_t)e * DDIM * IDIM + n0;

  // Double-buffer state: explicit cur/next pointers + LDS byte offsets.
  const float* curg = &ldsg[0][0];
  const float* curu = &ldsu[0][0];
  const float* nxtg = &ldsg[1][0];
  const float* nxtu = &ldsu[1][0];
  unsigned cur_og = (unsigned)(uintptr_t)curg;
  unsigned cur_ou = (unsigned)(uintptr_t)curu;
  unsigned nxt_og = (unsigned)(uintptr_t)nxtg;
  unsigned nxt_ou = (unsigned)(uintptr_t)nxtu;

  // Prologue: stage K-step 0.
  if (wid == 0) {
    tdm_load_2d(cur_og, wgE, IDIM, DDIM, 16, 32, IDIM);
    tdm_load_2d(cur_ou, wuE, IDIM, DDIM, 16, 32, IDIM);
    __builtin_amdgcn_s_wait_tensorcnt(0);
  }
  __syncthreads();

#pragma clang loop pipeline(disable)
#pragma unroll 1
  for (int kk = 0; kk < DDIM; kk += 32) {
    // Issue DMA for the next K-step into the other buffer (overlaps compute).
    if (wid == 0 && (kk + 32) < DDIM) {
      tdm_load_2d(nxt_og, wgE + (size_t)(kk + 32) * IDIM, IDIM, DDIM, 16, 32, IDIM);
      tdm_load_2d(nxt_ou, wuE + (size_t)(kk + 32) * IDIM, IDIM, DDIM, 16, 32, IDIM);
    }

    const int kb = kk + kadd;
    v16h bg = load_b_lds(curg, kadd, rlo);
    v16h bu = load_b_lds(curu, kadd, rlo);
#pragma unroll
    for (int mt = 0; mt < 4; ++mt) {
      v16h a = load_a_f32(x, tok[mt], kb, DDIM);
      accg[mt] = __builtin_amdgcn_wmma_f32_16x16x32_f16(
          false, a, false, bg, (short)0, accg[mt], false, false);
      accu[mt] = __builtin_amdgcn_wmma_f32_16x16x32_f16(
          false, a, false, bu, (short)0, accu[mt], false, false);
    }

    if (wid == 0) __builtin_amdgcn_s_wait_tensorcnt(0);
    __syncthreads();

    // Swap buffers.
    const float* tp;
    unsigned to;
    tp = curg; curg = nxtg; nxtg = tp;
    tp = curu; curu = nxtu; nxtu = tp;
    to = cur_og; cur_og = nxt_og; nxt_og = to;
    to = cur_ou; cur_ou = nxt_ou; nxt_ou = to;
  }

  // Epilogue: hidden = silu(gate) * up, stored f16 (WMMA-native for pass 2).
#pragma unroll
  for (int mt = 0; mt < 4; ++mt) {
#pragma unroll
    for (int j = 0; j < 8; ++j) {
      int p = tsw + mt * 16 + j + kadd;     // C/D row: M = j (+8 for hi lanes)
      if (p < te) {
        int t = ws[WS_PERM + p];
        float g = accg[mt][j], u = accu[mt][j];
        float h = (g / (1.0f + __expf(-g))) * u;
        hidden[(size_t)t * IDIM + n] = (_Float16)h;
      }
    }
  }
}

// ---------------------------------------------------------------------------
// Kernel 3: out = hidden · Wd[e].  Same structure, K = I, one weight matrix.
// ---------------------------------------------------------------------------
__global__ __launch_bounds__(128) void moe_gemm2(const _Float16* __restrict__ hidden,
                                                 const float* __restrict__ wd,
                                                 const int* __restrict__ ws,
                                                 float* __restrict__ out) {
  __shared__ float ldsd[2][32 * 16];

  const int nmb = ws[0];
  if ((int)blockIdx.y >= nmb) return;
  const int e    = ws[WS_MB_E + blockIdx.y];
  const int ts   = ws[WS_MB_S + blockIdx.y];
  const int te   = ws[WS_MB_END + blockIdx.y];
  const int lane = threadIdx.x & 31;
  const int rlo  = lane & 15;
  const int kadd = (lane >> 4) << 3;
  const int n0   = blockIdx.x * 16;
  const int n    = n0 + rlo;                  // D column
  const int wid  = __builtin_amdgcn_readfirstlane((int)(threadIdx.x >> 5));
  const int tsw  = ts + wid * 64;

  const int tok0 = ws[WS_PERM + ts];
  int tok[4];
#pragma unroll
  for (int mt = 0; mt < 4; ++mt) {
    int p = tsw + mt * 16 + rlo;
    tok[mt] = (p < te) ? ws[WS_PERM + p] : tok0;
  }

  v8f acc[4];
#pragma unroll
  for (int mt = 0; mt < 4; ++mt)
#pragma unroll
    for (int j = 0; j < 8; ++j) acc[mt][j] = 0.0f;

  const float* __restrict__ wdE = wd + (size_t)e * IDIM * DDIM + n0;

  const float* curd = &ldsd[0][0];
  const float* nxtd = &ldsd[1][0];
  unsigned cur_od = (unsigned)(uintptr_t)curd;
  unsigned nxt_od = (unsigned)(uintptr_t)nxtd;

  if (wid == 0) {
    tdm_load_2d(cur_od, wdE, DDIM, IDIM, 16, 32, DDIM);
    __builtin_amdgcn_s_wait_tensorcnt(0);
  }
  __syncthreads();

#pragma clang loop pipeline(disable)
#pragma unroll 1
  for (int kk = 0; kk < IDIM; kk += 32) {
    if (wid == 0 && (kk + 32) < IDIM) {
      tdm_load_2d(nxt_od, wdE + (size_t)(kk + 32) * DDIM, DDIM, IDIM, 16, 32, DDIM);
    }

    const int kb = kk + kadd;
    v16h bd = load_b_lds(curd, kadd, rlo);
#pragma unroll
    for (int mt = 0; mt < 4; ++mt) {
      const v8h* hp = (const v8h*)(hidden + (size_t)tok[mt] * IDIM + kb);
      v8h h0 = hp[0];   // K = kb .. kb+7
      v8h h1 = hp[2];   // K = kb+16 .. kb+23
      v16h a = __builtin_shufflevector(h0, h1, 0, 1, 2, 3, 4, 5, 6, 7,
                                       8, 9, 10, 11, 12, 13, 14, 15);
      acc[mt] = __builtin_amdgcn_wmma_f32_16x16x32_f16(
          false, a, false, bd, (short)0, acc[mt], false, false);
    }

    if (wid == 0) __builtin_amdgcn_s_wait_tensorcnt(0);
    __syncthreads();

    const float* tp = curd; curd = nxtd; nxtd = tp;
    unsigned to = cur_od; cur_od = nxt_od; nxt_od = to;
  }

#pragma unroll
  for (int mt = 0; mt < 4; ++mt) {
#pragma unroll
    for (int j = 0; j < 8; ++j) {
      int p = tsw + mt * 16 + j + kadd;
      if (p < te) {
        int t = ws[WS_PERM + p];
        out[(size_t)t * DDIM + n] = acc[mt][j];
      }
    }
  }
}

// ---------------------------------------------------------------------------
extern "C" void kernel_launch(void* const* d_in, const int* in_sizes, int n_in,
                              void* d_out, int out_size, void* d_ws, size_t ws_size,
                              hipStream_t stream) {
  const float* x   = (const float*)d_in[0];
  const int*   idx = (const int*)d_in[1];
  const float* wg  = (const float*)d_in[2];
  const float* wu  = (const float*)d_in[3];
  const float* wd  = (const float*)d_in[4];
  float* out = (float*)d_out;

  int* wsi = (int*)d_ws;
  _Float16* hidden = (_Float16*)((char*)d_ws + WS_HIDDEN_BYTE_OFF);

  moe_setup<<<1, 256, 0, stream>>>(idx, wsi);

  dim3 g1(IDIM / 16, MAXMB);  // (128, 16); inactive M-blocks exit early
  moe_gemm1<<<g1, 128, 0, stream>>>(x, wg, wu, wsi, hidden);

  dim3 g2(DDIM / 16, MAXMB);  // (64, 16)
  moe_gemm2<<<g2, 128, 0, stream>>>(hidden, wd, wsi, out);
}
